// LowrankApproximate2Layer_807453852458
// MI455X (gfx1250) — matrix operand order repaired and checked by reading, not compiled, not run on hardware
//
#include <hip/hip_runtime.h>
#include <hip/hip_bf16.h>

#define KNN 32

typedef __bf16 bf16;
typedef __attribute__((ext_vector_type(16))) __bf16 v16bf;
typedef __attribute__((ext_vector_type(8)))  __bf16 bf16x8;
typedef __attribute__((ext_vector_type(8)))  float  v8f;

// ---------------------------------------------------------------------------
// f32 -> (bf16 hi, bf16 lo) split conversion.  x ~= hi + lo with ~16 mantissa
// bits total, enabling emulated-f32 GEMM at bf16 WMMA rate (3 products).
// ---------------------------------------------------------------------------
__global__ void split_bf16_kernel(const float* __restrict__ in,
                                  bf16* __restrict__ hi, bf16* __restrict__ lo,
                                  int n) {
    int i = blockIdx.x * blockDim.x + threadIdx.x;
    if (i < n) {
        float v = in[i];
        bf16 h = (bf16)v;
        hi[i] = h;
        lo[i] = (bf16)(v - (float)h);
    }
}

// ---------------------------------------------------------------------------
// WMMA fragment loaders per CDNA5 ISA 7.12.2 (wave32, 16-bit operands).
// A matrix 16x32 (MxK): lane holds row M = lane&15;
//   element i -> K = k0 + 16*(i>=8) + 8*(lane>=16) + (i&7)
// B matrix 32x16 (KxN): lane holds col N = lane&15;
//   element i -> K = k0 + 16*(i>=8) + 8*(lane>=16) + (i&7)
// Both are two contiguous 16-byte loads per lane.
// ---------------------------------------------------------------------------
__device__ __forceinline__ v16bf load_a_frag(const bf16* __restrict__ rowBase,
                                             int k0, int lane) {
    int kk = k0 + ((lane >> 4) << 3);   // +8 for lanes 16..31
    union { struct { bf16x8 a, b; } s; v16bf v; } u;
    u.s.a = *reinterpret_cast<const bf16x8*>(rowBase + kk);
    u.s.b = *reinterpret_cast<const bf16x8*>(rowBase + kk + 16);
    return u.v;
}

__device__ __forceinline__ v16bf load_b_frag(const bf16* __restrict__ colBase,
                                             int k0, int lane) {
    int kk = k0 + ((lane >> 4) << 3);   // +8 for lanes 16..31
    union { struct { bf16x8 a, b; } s; v16bf v; } u;
    u.s.a = *reinterpret_cast<const bf16x8*>(colBase + kk);
    u.s.b = *reinterpret_cast<const bf16x8*>(colBase + kk + 16);
    return u.v;
}

// ---------------------------------------------------------------------------
// NT GEMM tile: one wave computes a 16(M) x 64(N) f32 tile of
//   C[m,n] = sum_k A[m,k] * B[n,k]      (A row-major lda, B row-major ldb)
// using split-bf16 emulation: A = Ah+Al, B = Bh+Bl,
//   C ~= Ah*Bh + Ah*Bl + Al*Bh   (Al*Bl dropped, below f32 ulp)
// 3 x v_wmma_f32_16x16x32_bf16 per K-step per 16x16 subtile.
// ---------------------------------------------------------------------------
__device__ __forceinline__ void compute_tile(
    const bf16* __restrict__ Ah, const bf16* __restrict__ Al, int lda,
    const bf16* __restrict__ Bh, const bf16* __restrict__ Bl, int ldb,
    int mTile, int nTile, int K, int lane, v8f acc[4]) {

    const bf16* arh = Ah + (size_t)(mTile + (lane & 15)) * lda;
    const bf16* arl = Al + (size_t)(mTile + (lane & 15)) * lda;

    for (int k0 = 0; k0 < K; k0 += 32) {
        v16bf fah = load_a_frag(arh, k0, lane);
        v16bf fal = load_a_frag(arl, k0, lane);
#pragma unroll
        for (int s = 0; s < 4; ++s) {
            const bf16* bch = Bh + (size_t)(nTile + s * 16 + (lane & 15)) * ldb;
            const bf16* bcl = Bl + (size_t)(nTile + s * 16 + (lane & 15)) * ldb;
            v16bf fbh = load_b_frag(bch, k0, lane);
            v16bf fbl = load_b_frag(bcl, k0, lane);
            acc[s] = __builtin_amdgcn_wmma_f32_16x16x32_bf16(
                false, fah, false, fbh, (short)0, acc[s], false, false);
            acc[s] = __builtin_amdgcn_wmma_f32_16x16x32_bf16(
                false, fah, false, fbl, (short)0, acc[s], false, false);
            acc[s] = __builtin_amdgcn_wmma_f32_16x16x32_bf16(
                false, fal, false, fbh, (short)0, acc[s], false, false);
        }
    }
}

// ---------------------------------------------------------------------------
// Step 0: W'[e, h*1024 + sel*512 + k] = sum_d Wq[e, h*1024+sel*512+d] * key[h,k,d]
// 8 batched GEMMs (1024 x 512 x 512).  Output written TRANSPOSED and hi/lo
// split: Wt[j][e], so step 1's B-fragments are contiguous.
// Block = 256 threads (8 waves); wave w -> rows [mTile, mTile+16).
// Grid = (1024/128, 512/64, 8 batches).
// ---------------------------------------------------------------------------
__global__ void gemm_wt_kernel(const bf16* __restrict__ Wqh, const bf16* __restrict__ Wql,
                               const bf16* __restrict__ K1h, const bf16* __restrict__ K1l,
                               const bf16* __restrict__ K2h, const bf16* __restrict__ K2l,
                               bf16* __restrict__ Wth, bf16* __restrict__ Wtl) {
    int lane = threadIdx.x & 31;
    int wv   = threadIdx.x >> 5;
    int b    = blockIdx.z;
    int h = b >> 1, sel = b & 1;
    int mTile = blockIdx.x * 128 + wv * 16;   // e
    int nTile = blockIdx.y * 64;              // k
    int off = h * 1024 + sel * 512;

    const bf16* Ah = Wqh + off;                         // lda = 4096
    const bf16* Al = Wql + off;
    const bf16* Bh = (sel ? K2h : K1h) + (size_t)h * 512 * 512;   // ldb = 512
    const bf16* Bl = (sel ? K2l : K1l) + (size_t)h * 512 * 512;

    v8f acc[4] = {};
    compute_tile(Ah, Al, 4096, Bh, Bl, 512, mTile, nTile, 512, lane, acc);

    int mBase = mTile + ((lane >> 4) << 3);   // C layout: VGPR r -> M = mBase + r
    int nCol  = (lane & 15);
#pragma unroll
    for (int s = 0; s < 4; ++s) {
        int jg = off + nTile + s * 16 + nCol; // global j (column of W')
#pragma unroll
        for (int r = 0; r < 8; ++r) {
            float v = acc[s][r];
            int e = mBase + r;
            bf16 hh = (bf16)v;
            Wth[(size_t)jg * 1024 + e] = hh;
            Wtl[(size_t)jg * 1024 + e] = (bf16)(v - (float)hh);
        }
    }
}

// ---------------------------------------------------------------------------
// Step 1: S[n, j] = sum_e x[n,e] * Wt[j,e]     (2048 x 4096 x 1024)
// Grid = (2048/128, 4096/64), 256 threads.
// ---------------------------------------------------------------------------
__global__ void gemm_s_kernel(const bf16* __restrict__ Xh, const bf16* __restrict__ Xl,
                              const bf16* __restrict__ Wth, const bf16* __restrict__ Wtl,
                              float* __restrict__ S) {
    int lane = threadIdx.x & 31;
    int wv   = threadIdx.x >> 5;
    int mTile = blockIdx.x * 128 + wv * 16;   // n (x row)
    int nTile = blockIdx.y * 64;              // j

    v8f acc[4] = {};
    compute_tile(Xh, Xl, 1024, Wth, Wtl, 1024, mTile, nTile, 1024, lane, acc);

    int mBase = mTile + ((lane >> 4) << 3);
    int nCol  = (lane & 15);
#pragma unroll
    for (int s = 0; s < 4; ++s) {
#pragma unroll
        for (int r = 0; r < 8; ++r) {
            S[(size_t)(mBase + r) * 4096 + nTile + s * 16 + nCol] = acc[s][r];
        }
    }
}

// ---------------------------------------------------------------------------
// Two-stage top-32 per (n,h): one wave32 per task.
// Stage A: top-32 of s1[512] and s2[512] (iterative wave argmax, JAX-stable
//          tie-break = lowest original index).
// Stage B: top-32 of the 32x32 pair sums, tie-break = lowest comb position
//          pos = i*32 + j (lane = rank i of t1).
// Emits ind = i1*512 + i2 and w = relu(t1v[i]+t2v[j]).
// ---------------------------------------------------------------------------
__global__ void topk_kernel(const float* __restrict__ S,
                            int* __restrict__ ind_out, float* __restrict__ w_out) {
    int wid  = (blockIdx.x * blockDim.x + threadIdx.x) >> 5;
    int lane = threadIdx.x & 31;
    int slot = threadIdx.x >> 5;
    int n = wid >> 2, h = wid & 3;

    const float* s1 = S + (size_t)n * 4096 + h * 1024;
    const float* s2 = s1 + 512;

    float v1[16], v2[16];
#pragma unroll
    for (int j = 0; j < 16; ++j) {
        v1[j] = s1[j * 32 + lane];      // element index e = j*32 + lane
        v2[j] = s2[j * 32 + lane];
    }

    __shared__ float t1v_s[8][32], t2v_s[8][32];
    __shared__ int   t1i_s[8][32], t2i_s[8][32];

    const float NEG_INF = -__builtin_inff();

#pragma unroll 1
    for (int list = 0; list < 2; ++list) {
        float* vv = list ? v2 : v1;
        float* tv = list ? t2v_s[slot] : t1v_s[slot];
        int*   ti = list ? t2i_s[slot] : t1i_s[slot];
#pragma unroll 1
        for (int r = 0; r < KNN; ++r) {
            float best = NEG_INF; int bidx = 0x7fffffff;
#pragma unroll
            for (int j = 0; j < 16; ++j) {
                if (vv[j] > best) { best = vv[j]; bidx = j * 32 + lane; }
            }
            float rv = best; int ri = bidx;
#pragma unroll
            for (int m = 16; m >= 1; m >>= 1) {
                float ov = __shfl_xor(rv, m, 32);
                int   oi = __shfl_xor(ri, m, 32);
                if (ov > rv || (ov == rv && oi < ri)) { rv = ov; ri = oi; }
            }
            if ((ri & 31) == lane) vv[ri >> 5] = NEG_INF;   // owner retires it
            if (lane == 0) { tv[r] = rv; ti[r] = ri; }
        }
    }
    __syncthreads();

    float a = t1v_s[slot][lane];       // lane == rank i in t1
    unsigned used = 0u;                // per-lane: which j already taken
#pragma unroll 1
    for (int r = 0; r < KNN; ++r) {
        float best = NEG_INF; int bpos = 0x7fffffff;
        for (int j = 0; j < 32; ++j) {
            if (!((used >> j) & 1u)) {
                float c = a + t2v_s[slot][j];
                int pos = lane * 32 + j;
                if (c > best || (c == best && pos < bpos)) { best = c; bpos = pos; }
            }
        }
        float rv = best; int rp = bpos;
#pragma unroll
        for (int m = 16; m >= 1; m >>= 1) {
            float ov = __shfl_xor(rv, m, 32);
            int   op = __shfl_xor(rp, m, 32);
            if (ov > rv || (ov == rv && op < rp)) { rv = ov; rp = op; }
        }
        int wi = rp >> 5, wj = rp & 31;
        if (lane == wi) used |= (1u << wj);
        if (lane == 0) {
            int index = t1i_s[slot][wi] * 512 + t2i_s[slot][wj];
            size_t o = ((size_t)n * 4 + h) * 32 + r;
            ind_out[o] = index;
            w_out[o]   = rv > 0.f ? rv : 0.f;
        }
    }
}

// ---------------------------------------------------------------------------
// out[n, :] = sum_{k<128} w[n,k] * values[ind[n,k], :]
// One block per n; thread t owns float4 columns [4t, 4t+4).  ~1 GB of gather
// traffic -> this step is the HBM roofline of the whole problem.
// ---------------------------------------------------------------------------
__global__ void gather_kernel(const float* __restrict__ values,
                              const int* __restrict__ ind,
                              const float* __restrict__ w,
                              float* __restrict__ out) {
    int n = blockIdx.x;
    __shared__ float ws[128];
    __shared__ int   is[128];
    if (threadIdx.x < 128) {
        ws[threadIdx.x] = w[(size_t)n * 128 + threadIdx.x];
        is[threadIdx.x] = ind[(size_t)n * 128 + threadIdx.x];
    }
    __syncthreads();

    int c = threadIdx.x * 4;
    float4 acc = make_float4(0.f, 0.f, 0.f, 0.f);
#pragma unroll 4
    for (int k = 0; k < 128; ++k) {
        float wv = ws[k];
        if (wv != 0.f) {
            const float4 v = *reinterpret_cast<const float4*>(
                values + (size_t)is[k] * 1024 + c);
            acc.x += wv * v.x; acc.y += wv * v.y;
            acc.z += wv * v.z; acc.w += wv * v.w;
        }
    }
    *reinterpret_cast<float4*>(out + (size_t)n * 1024 + c) = acc;
}

// ---------------------------------------------------------------------------
extern "C" void kernel_launch(void* const* d_in, const int* in_sizes, int n_in,
                              void* d_out, int out_size, void* d_ws, size_t ws_size,
                              hipStream_t stream) {
    const float* x    = (const float*)d_in[0];   // 2048 x 1024
    const float* Wq   = (const float*)d_in[1];   // 1024 x 4096
    const float* key1 = (const float*)d_in[2];   // 4 x 512 x 512
    const float* key2 = (const float*)d_in[3];   // 4 x 512 x 512
    const float* vals = (const float*)d_in[4];   // 262144 x 1024
    float* out = (float*)d_out;                  // 2048 x 1024

    // Workspace carve-out (256B aligned regions)
    size_t o = 0;
    auto carve = [&](size_t bytes) -> char* {
        char* p = (char*)d_ws + o;
        o += (bytes + 255) & ~(size_t)255;
        return p;
    };
    const size_t NWQ = 1024u * 4096u, NK = 4u * 512u * 512u, NX = 2048u * 1024u;
    bf16* Wqh = (bf16*)carve(NWQ * 2);  bf16* Wql = (bf16*)carve(NWQ * 2);
    bf16* K1h = (bf16*)carve(NK * 2);   bf16* K1l = (bf16*)carve(NK * 2);
    bf16* K2h = (bf16*)carve(NK * 2);   bf16* K2l = (bf16*)carve(NK * 2);
    bf16* Xh  = (bf16*)carve(NX * 2);   bf16* Xl  = (bf16*)carve(NX * 2);
    bf16* Wth = (bf16*)carve(NWQ * 2);  bf16* Wtl = (bf16*)carve(NWQ * 2); // 4096 x 1024
    float* S  = (float*)carve(2048u * 4096u * 4);                          // 2048 x 4096
    int*   IB = (int*)carve(2048u * 128u * 4);
    float* WB = (float*)carve(2048u * 128u * 4);
    (void)ws_size; (void)in_sizes; (void)n_in; (void)out_size;

    // 1) split f32 inputs into bf16 hi/lo
    split_bf16_kernel<<<(int)((NWQ + 255) / 256), 256, 0, stream>>>(Wq, Wqh, Wql, (int)NWQ);
    split_bf16_kernel<<<(int)((NK  + 255) / 256), 256, 0, stream>>>(key1, K1h, K1l, (int)NK);
    split_bf16_kernel<<<(int)((NK  + 255) / 256), 256, 0, stream>>>(key2, K2h, K2l, (int)NK);
    split_bf16_kernel<<<(int)((NX  + 255) / 256), 256, 0, stream>>>(x, Xh, Xl, (int)NX);

    // 2) W' = fold keys into Wq (8 batched 1024x512x512 GEMMs), store transposed+split
    gemm_wt_kernel<<<dim3(8, 8, 8), 256, 0, stream>>>(Wqh, Wql, K1h, K1l, K2h, K2l, Wth, Wtl);

    // 3) S = x @ W'   (2048 x 4096 x 1024)
    gemm_s_kernel<<<dim3(16, 64), 256, 0, stream>>>(Xh, Xl, Wth, Wtl, S);

    // 4) two-stage top-32 per (n,h): 8192 waves
    topk_kernel<<<1024, 256, 0, stream>>>(S, IB, WB);

    // 5) weighted gather of values rows
    gather_kernel<<<2048, 256, 0, stream>>>(vals, IB, WB, out);
}
